// _NonLocalBlock3_75909251989706
// MI455X (gfx1250) — matrix operand order repaired and checked by reading, not compiled, not run on hardware
//
#include <hip/hip_runtime.h>

typedef __attribute__((ext_vector_type(16))) __bf16 v16bf;
typedef __attribute__((ext_vector_type(8)))  float  v8f;

#define B_  2
#define C_  128
#define IC_ 64
#define W_  32
#define H_  32
#define Z_  128
#define NZ_ 64
#define SC_ (W_*H_*Z_)        /* 131072: stride of channel dim */
#define SW_ (H_*Z_)           /* 4096                          */
#define SB_ (C_*W_*H_*Z_)     /* 16777216                      */

// ---- workspace byte offsets (prep kernel fills these) ----
#define WSU   0        // u[128] f32 : cat_w[:ic]^T @ theta_w
#define WSCT  512      // ct f32     : cat_w[:ic] . theta_b
#define WSSC  768      // sc[128] f32: bn_g/sqrt(bn_v+eps)
#define WSSH  1280     // sh[128] f32: (W_b-bn_m)*sc + bn_b
#define WSGW  2048     // g_w   bf16 [64][128]
#define WSPW  18432    // phi_w bf16 [64][128]
#define WSWW  34816    // W_w   bf16 [128][64]

static __device__ __forceinline__ unsigned short f2bf(float f) {
  unsigned int u = __float_as_uint(f);
  u += 0x7FFFu + ((u >> 16) & 1u);          // round-to-nearest-even
  return (unsigned short)(u >> 16);
}
// packed pair conversion: one v_cvt_pk_bf16_f32 when available
static __device__ __forceinline__ unsigned int f2bf2(float a, float b) {
#if __has_builtin(__builtin_amdgcn_cvt_pk_bf16_f32)
  auto r = __builtin_amdgcn_cvt_pk_bf16_f32(a, b);
  union { decltype(r) v; unsigned int u; } cvt;
  cvt.v = r;
  return cvt.u;
#else
  return (unsigned int)f2bf(a) | ((unsigned int)f2bf(b) << 16);
#endif
}
static __device__ __forceinline__ float bf2f(unsigned short h) {
  return __uint_as_float(((unsigned int)h) << 16);
}

union AFrag { v16bf v; uint4 q[2]; unsigned int u[8]; unsigned short s[16]; };
union CFrag { v8f   v; float f[8]; };
union Pk8   { uint4 q; unsigned int u[4]; unsigned short s[8]; };

// A operand (MxK bf16 16x32), row-major source; 16B-aligned rows required.
// lane&15 = M row; elems 0..15 = K k0+16*(lane>>4)+e (contiguous)
static __device__ __forceinline__ v16bf load_a_rm(const unsigned short* M, int pitch,
                                                  int m0, int k0, int lane) {
  const unsigned short* r = M + (m0 + (lane & 15)) * pitch + k0 + ((lane >> 4) << 4);
  AFrag a;
  a.q[0] = *(const uint4*)(r);
  a.q[1] = *(const uint4*)(r + 8);
  return a.v;
}

// B operand from pre-staged fragment layout: [tile][lane][16 bf16 contiguous]
static __device__ __forceinline__ v16bf load_b_staged(const unsigned short* S,
                                                      int tile, int lane) {
  const unsigned short* p = S + (tile * 32 + lane) * 16;
  AFrag b;
  b.q[0] = *(const uint4*)(p);
  b.q[1] = *(const uint4*)(p + 8);
  return b.v;
}

// element (k,n) of a (K x N) matrix -> index in staged fragment layout.
// ntiles = N/16. lanes 0-15 hold K k0..k0+15 (e=kin&15), lanes 16-31 K+16.
static __device__ __forceinline__ int bstage_idx(int k, int n, int ntiles) {
  int kt = k >> 5, kin = k & 31;
  int nt = n >> 4, nin = n & 15;
  int lane = nin + ((kin >> 4) << 4);
  return ((kt * ntiles + nt) * 32 + lane) * 16 + (kin & 15);
}

#define WMMA(acc, a, b) \
  (acc) = __builtin_amdgcn_wmma_f32_16x16x32_bf16(false, (a), false, (b), (short)0, (acc), false, false)

// -------- prep: bf16 weight conversion + theta collapse + BN fold --------
__global__ __launch_bounds__(256) void nlb_prep(
    const float* __restrict__ g_w, const float* __restrict__ theta_w,
    const float* __restrict__ theta_b, const float* __restrict__ phi_w,
    const float* __restrict__ cat_w, const float* __restrict__ W_w,
    const float* __restrict__ W_b, const float* __restrict__ bn_g,
    const float* __restrict__ bn_b, const float* __restrict__ bn_m,
    const float* __restrict__ bn_v, unsigned char* __restrict__ ws) {
  float* u  = (float*)(ws + WSU);
  float* ct = (float*)(ws + WSCT);
  float* sc = (float*)(ws + WSSC);
  float* sh = (float*)(ws + WSSH);
  unsigned short* gwb = (unsigned short*)(ws + WSGW);
  unsigned short* pwb = (unsigned short*)(ws + WSPW);
  unsigned short* wwb = (unsigned short*)(ws + WSWW);
  const int t = threadIdx.x;
  for (int i = t; i < IC_ * C_; i += 256) {
    gwb[i] = f2bf(g_w[i]);
    pwb[i] = f2bf(phi_w[i]);
    wwb[i] = f2bf(W_w[i]);          // C_*IC_ == IC_*C_ element count
  }
  if (t < C_) {
    float s = 0.f;
    for (int ic = 0; ic < IC_; ++ic) s += cat_w[ic] * theta_w[ic * C_ + t];
    u[t] = s;
    float scale = bn_g[t] * rsqrtf(bn_v[t] + 1e-5f);
    sc[t] = scale;
    sh[t] = (W_b[t] - bn_m[t]) * scale + bn_b[t];
  }
  if (t == 0) {
    float s = 0.f;
    for (int ic = 0; ic < IC_; ++ic) s += cat_w[ic] * theta_b[ic];
    *ct = s;
  }
}

// -------- fused per-site kernel: one workgroup per (b,w,h) --------
__global__ __launch_bounds__(256) void nlb_main(
    const float* __restrict__ x, const float* __restrict__ g_b,
    const float* __restrict__ phi_b, const float* __restrict__ cat_w,
    const unsigned char* __restrict__ ws, float* __restrict__ out) {
  // 60160 B of LDS, phase-aliased
  __shared__ __align__(16) unsigned char smem[60160];
  unsigned short* Xs  = (unsigned short*)smem;            // staged X (128k x 128n), 32 tiles
  unsigned short* Ys  = (unsigned short*)smem;            // staged y^T (64k x 128n), reuse
  unsigned short* St  = (unsigned short*)(smem + 32768);  // conv out transposed [128 z][72 ic]
  unsigned short* Fbf = St;                               // f row-major [128][72] (reuse)
  unsigned short* Gxs = (unsigned short*)(smem + 51200);  // staged Gx (64k x 64n), 8 tiles
  float* aV  = (float*)(smem + 59392);                    // [128]
  float* bbV = (float*)(smem + 59904);                    // [64]

  const float* u  = (const float*)(ws + WSU);
  const float* sc = (const float*)(ws + WSSC);
  const float* sh = (const float*)(ws + WSSH);
  const unsigned short* Gw = (const unsigned short*)(ws + WSGW);
  const unsigned short* Pw = (const unsigned short*)(ws + WSPW);
  const unsigned short* Ww = (const unsigned short*)(ws + WSWW);

  const int tid  = threadIdx.x;
  const int lane = tid & 31;
  const int wv   = tid >> 5;       // 8 waves
  const int hi8  = (lane >> 4) << 3;

  const int n  = blockIdx.x;
  const int b  = n >> 10;
  const int wI = (n & 1023) >> 5;
  const int hI = n & 31;
  const long gbase = (long)b * SB_ + (long)wI * SW_ + (long)hI * Z_;
  const float* xb = x + gbase;
  float* ob = out + gbase;

  // ---- phase 0: gather x(n,C,Z) straight into staged B-fragment layout ----
  // 32 tiles (kt=c/32, nt=z/16) x 32 lanes; each thread fills 4 fragment-lanes.
  for (int fi = tid; fi < 32 * 32; fi += 256) {
    const int tile = fi >> 5;            // kt*8 + nt
    const int ln   = fi & 31;
    const int kt = tile >> 3, nt = tile & 7;
    const int z  = nt * 16 + (ln & 15);
    const int c0 = kt * 32 + ((ln >> 4) << 4);
    const float* src = xb + (long)c0 * SC_ + z;
    AFrag fr;
#pragma unroll
    for (int e = 0; e < 16; e += 2)
      fr.u[e >> 1] = f2bf2(src[(long)e * SC_], src[(long)(e + 1) * SC_]);
    unsigned short* dst = Xs + (tile * 32 + ln) * 16;
    *(uint4*)(dst)     = fr.q[0];
    *(uint4*)(dst + 8) = fr.q[1];
  }
  __syncthreads();

  // ---- phases 1 & 3: g/phi convs: W(64ic x 128c) @ X(128c x 128z) ----
  for (int which = 0; which < 2; ++which) {
    const unsigned short* Wt = which ? Pw : Gw;
    const float* bias = which ? phi_b : g_b;
    const int t0 = wv * 4;               // 4 tiles/wave, fixed M-tile -> hoist A
    const int m0 = (t0 >> 3) * 16;
    v16bf a0 = load_a_rm(Wt, C_, m0,  0, lane);
    v16bf a1 = load_a_rm(Wt, C_, m0, 32, lane);
    v16bf a2 = load_a_rm(Wt, C_, m0, 64, lane);
    v16bf a3 = load_a_rm(Wt, C_, m0, 96, lane);
    for (int i = 0; i < 4; ++i) {
      const int nt = (t0 + i) & 7;
      CFrag acc;
#pragma unroll
      for (int r = 0; r < 8; ++r) acc.f[r] = bias[m0 + r + hi8];
      WMMA(acc.v, a0, load_b_staged(Xs, 0 * 8 + nt, lane));
      WMMA(acc.v, a1, load_b_staged(Xs, 1 * 8 + nt, lane));
      WMMA(acc.v, a2, load_b_staged(Xs, 2 * 8 + nt, lane));
      WMMA(acc.v, a3, load_b_staged(Xs, 3 * 8 + nt, lane));
      // transposed store: St[z][ic]; lane's 8 rows are contiguous -> one b128
      const int zc = nt * 16 + (lane & 15);
      Pk8 pk;
#pragma unroll
      for (int r = 0; r < 4; ++r) pk.u[r] = f2bf2(acc.f[2 * r], acc.f[2 * r + 1]);
      *(uint4*)(St + zc * 72 + m0 + hi8) = pk.q;
    }
    __syncthreads();
    if (which == 0) {
      // ---- phase 2: z-pair max-pool g, write staged (K=j 64, N=ic 64) ----
      for (int i = tid; i < NZ_ * IC_; i += 256) {
        const int j = i >> 6, ic = i & 63;
        const float v0 = bf2f(St[(2 * j) * 72 + ic]);
        const float v1 = bf2f(St[(2 * j + 1) * 72 + ic]);
        Gxs[bstage_idx(j, ic, 4)] = f2bf(fmaxf(v0, v1));
      }
      __syncthreads();
    }
  }

  // ---- phase 4: bb[j] from pooled phi; a[z] via collapsed theta ----
  if (tid < NZ_) {
    const int j = tid;
    const float* cw2 = cat_w + IC_;
    float s = 0.f;
    for (int ic = 0; ic < IC_; ++ic) {
      const float v0 = bf2f(St[(2 * j) * 72 + ic]);
      const float v1 = bf2f(St[(2 * j + 1) * 72 + ic]);
      s += cw2[ic] * fmaxf(v0, v1);
    }
    bbV[j] = s;
  } else if (tid < 64 + Z_) {
    const int z = tid - 64;
    const int nt = z >> 4, nin = z & 15;
    float s = *(const float*)(ws + WSCT);
    for (int kt = 0; kt < 4; ++kt) {
      const unsigned short* base = Xs + ((kt * 8 + nt) * 32) * 16;
      const unsigned short* plo = base + nin * 16;
      const unsigned short* phh = base + (16 + nin) * 16;
#pragma unroll
      for (int e = 0; e < 16; ++e) {
        s += u[kt * 32 + e]      * bf2f(plo[e]);
        s += u[kt * 32 + 16 + e] * bf2f(phh[e]);
      }
    }
    aV[z] = s;
  }
  __syncthreads();

  // ---- phase 5: f = relu(a_i + b_j)/nz (overwrites conv scratch region) ----
  for (int i = tid; i < Z_ * (NZ_ / 4); i += 256) {
    const int z = i >> 4, j4 = (i & 15) << 2;
    const float az = aV[z];
    Pk8 pk;
#pragma unroll
    for (int j = 0; j < 4; j += 2)
      pk.u[j >> 1] = f2bf2(fmaxf(az + bbV[j4 + j], 0.f) * (1.0f / NZ_),
                           fmaxf(az + bbV[j4 + j + 1], 0.f) * (1.0f / NZ_));
    *(uint2*)(Fbf + z * 72 + j4) = *(const uint2*)&pk.q;
  }
  __syncthreads();

  // ---- phase 6: y = f(128z x 64j) @ Gx(64j x 64ic); stage y^T (K=ic,N=z) ----
  {
    // prefetch this wave's phase-7 residual rows (c = wv*16..+15) into near cache
    {
      const int c = wv * 16 + (lane >> 2);
      const float* pf = xb + (long)c * SC_ + (lane & 3) * 32;
      __builtin_prefetch(pf, 0, 0);
      __builtin_prefetch(pf + 8L * SC_, 0, 0);
    }
    const int m0 = wv * 16;              // z tile (one per wave)
    v16bf a0 = load_a_rm(Fbf, 72, m0,  0, lane);
    v16bf a1 = load_a_rm(Fbf, 72, m0, 32, lane);
    for (int nt = 0; nt < 4; ++nt) {
      CFrag acc;
#pragma unroll
      for (int r = 0; r < 8; ++r) acc.f[r] = 0.f;
      WMMA(acc.v, a0, load_b_staged(Gxs, 0 * 4 + nt, lane));
      WMMA(acc.v, a1, load_b_staged(Gxs, 1 * 4 + nt, lane));
      const int icc = nt * 16 + (lane & 15);
#pragma unroll
      for (int r = 0; r < 8; ++r)
        Ys[bstage_idx(icc, m0 + r + hi8, 8)] = f2bf(acc.f[r]);
    }
  }
  __syncthreads();

  // ---- phase 7: Wy = W_w(128c x 64ic) @ y^T(64ic x 128z); BN + residual ----
  {
    const int m0 = wv * 16;              // c tile (one per wave)
    v16bf a0 = load_a_rm(Ww, IC_, m0,  0, lane);
    v16bf a1 = load_a_rm(Ww, IC_, m0, 32, lane);
    for (int nt = 0; nt < 8; ++nt) {
      CFrag acc;
#pragma unroll
      for (int r = 0; r < 8; ++r) acc.f[r] = 0.f;
      WMMA(acc.v, a0, load_b_staged(Ys, 0 * 8 + nt, lane));
      WMMA(acc.v, a1, load_b_staged(Ys, 1 * 8 + nt, lane));
      const int z = nt * 16 + (lane & 15);
#pragma unroll
      for (int r = 0; r < 8; ++r) {
        const int c = m0 + r + hi8;
        const long off = (long)c * SC_ + z;
        ob[off] = acc.f[r] * sc[c] + sh[c] + xb[off];
      }
    }
  }
}

extern "C" void kernel_launch(void* const* d_in, const int* in_sizes, int n_in,
                              void* d_out, int out_size, void* d_ws, size_t ws_size,
                              hipStream_t stream) {
  const float* x       = (const float*)d_in[0];
  const float* g_w     = (const float*)d_in[1];
  const float* g_b     = (const float*)d_in[2];
  const float* theta_w = (const float*)d_in[3];
  const float* theta_b = (const float*)d_in[4];
  const float* phi_w   = (const float*)d_in[5];
  const float* phi_b   = (const float*)d_in[6];
  const float* cat_w   = (const float*)d_in[7];
  const float* W_w     = (const float*)d_in[8];
  const float* W_b     = (const float*)d_in[9];
  const float* bn_g    = (const float*)d_in[10];
  const float* bn_b    = (const float*)d_in[11];
  const float* bn_m    = (const float*)d_in[12];
  const float* bn_v    = (const float*)d_in[13];
  unsigned char* ws = (unsigned char*)d_ws;

  nlb_prep<<<1, 256, 0, stream>>>(g_w, theta_w, theta_b, phi_w, cat_w,
                                  W_w, W_b, bn_g, bn_b, bn_m, bn_v, ws);
  nlb_main<<<B_ * W_ * H_, 256, 0, stream>>>(x, g_b, phi_b, cat_w, ws,
                                             (float*)d_out);
}
